// GC_LSTM_20624432956159
// MI455X (gfx1250) — matrix operand order repaired and checked by reading, not compiled, not run on hardware
//
#include <hip/hip_runtime.h>

typedef __attribute__((ext_vector_type(16))) _Float16 v16h;
typedef __attribute__((ext_vector_type(8)))  float    v8f;
typedef __attribute__((ext_vector_type(4)))  int      v4i;

#define HID 32
#define Bsz 32
#define Tsz 16
#define Nsz 2048
#define ROWS (Bsz*Nsz)
#define WAVES_PER_BLOCK 8
#define ROWS_PER_WAVE 16
#define ROWS_PER_BLOCK (WAVES_PER_BLOCK*ROWS_PER_WAVE)

static __device__ __forceinline__ float sigm(float x){ return 1.0f/(1.0f+__expf(-x)); }
static __device__ __forceinline__ float tanhfast(float x){ return 2.0f*sigm(2.0f*x)-1.0f; }

// load 16 f16 from two 16-byte LDS chunks into a v16h operand
static __device__ __forceinline__ v16h ldv16(const _Float16* p0, const _Float16* p1){
  v16h r;
  ((v4i*)&r)[0] = *(const v4i*)p0;
  ((v4i*)&r)[1] = *(const v4i*)p1;
  return r;
}

__global__ __launch_bounds__(256) void gc_lstm_kernel(
    const float* __restrict__ x,
    const float* __restrict__ W_in, const float* __restrict__ b_in,
    const float* __restrict__ W_gc, const float* __restrict__ b_gc,
    const float* __restrict__ W_x2h, const float* __restrict__ b_x2h,
    const float* __restrict__ W_h2h, const float* __restrict__ b_h2h,
    const float* __restrict__ W_out, const float* __restrict__ b_out,
    float* __restrict__ out)
{
  // weights as f16 [N][K] row-major == original torch layout; biases f32
  __shared__ __align__(16) _Float16 sWx2h[128*64];
  __shared__ __align__(16) _Float16 sWh2h[128*32];
  __shared__ __align__(16) _Float16 sWgc [32*32];
  __shared__ float sBG[128];
  __shared__ float sBin[32], sBgc[32], sWin[32], sWout[32];
  // per-wave staging: xc = [xf | xg] (16x64 f16), h (16x32 f16), xn (16 f32)
  __shared__ __align__(16) _Float16 sXC[WAVES_PER_BLOCK][16*64];
  __shared__ __align__(16) _Float16 sH [WAVES_PER_BLOCK][16*32];
  __shared__ float sXN[WAVES_PER_BLOCK][16];

  const int tid = threadIdx.x;
  for (int i = tid; i < 128*64; i += 256) sWx2h[i] = (_Float16)W_x2h[i];
  for (int i = tid; i < 128*32; i += 256) sWh2h[i] = (_Float16)W_h2h[i];
  for (int i = tid; i < 32*32;  i += 256) sWgc[i]  = (_Float16)W_gc[i];
  if (tid < 128) sBG[tid] = b_x2h[tid] + b_h2h[tid];
  if (tid < 32) { sBin[tid]=b_in[tid]; sBgc[tid]=b_gc[tid]; sWin[tid]=W_in[tid]; sWout[tid]=W_out[tid]; }
  __syncthreads();

  const int w    = tid >> 5;
  const int lane = tid & 31;
  const int half = lane >> 4;   // which 16-lane half of the wave32
  const int lm   = lane & 15;
  const float bout = b_out[0];

  const int row  = blockIdx.x * ROWS_PER_BLOCK + w * ROWS_PER_WAVE + lm;
  const int bidx = row >> 11;          // N = 2048
  const int nidx = row & (Nsz - 1);

  // LSTM cell state in registers: hid = lm + 16*nt, row M = v + 8*half
  v8f cst0, cst1;
  #pragma unroll
  for (int v=0; v<8; ++v) { cst0[v]=0.f; cst1[v]=0.f; }

  // zero h staging (h0 = 0)
  for (int i = lane; i < 16*32/2; i += 32) ((unsigned int*)&sH[w][0])[i] = 0u;

  float xnv = x[bidx*(Tsz*Nsz) + nidx];   // x[:, 0, :] only input ever read

  _Float16* xcw = &sXC[w][0];
  _Float16* hw  = &sH[w][0];

  asm volatile("" ::: "memory");

  for (int t = 0; t < Tsz; ++t)
  {
    // ---- xf = relu(xn*W_in + b_in): lane fills row lm, cols 16*half..+15 ----
    __align__(16) unsigned int dw[8];
    #pragma unroll
    for (int jj=0; jj<8; ++jj){
      int h = 16*half + 2*jj;
      float v0 = fmaxf(fmaf(xnv, sWin[h],   sBin[h]),   0.f);
      float v1 = fmaxf(fmaf(xnv, sWin[h+1], sBin[h+1]), 0.f);
      union { _Float16 hh[2]; unsigned int u; } cv;
      cv.hh[0]=(_Float16)v0; cv.hh[1]=(_Float16)v1;
      dw[jj]=cv.u;
    }
    {
      v4i* dst = (v4i*)&xcw[lm*64 + 16*half];
      dst[0] = *(const v4i*)&dw[0];
      dst[1] = *(const v4i*)&dw[4];
    }
    asm volatile("" ::: "memory");

    // A operand: xf tile (= xc k-tile 0), A layout: row lm, K = 16*half + {0..15}
    v16h a0 = ldv16(&xcw[lm*64 +      16*half], &xcw[lm*64 + 16*half + 8]);

    // ---- xg = relu(xf @ Wgc^T + b_gc) -> xc cols 32..63 ----
    #pragma unroll
    for (int j=0; j<2; ++j){
      v16h bg = ldv16(&sWgc[(16*j+lm)*32 + 16*half], &sWgc[(16*j+lm)*32 + 16*half + 8]);
      float cb = sBgc[16*j+lm];
      v8f c = {cb,cb,cb,cb,cb,cb,cb,cb};
      c = __builtin_amdgcn_wmma_f32_16x16x32_f16(false, a0, false, bg, (short)0, c, false, false);
      #pragma unroll
      for (int v=0; v<8; ++v)
        xcw[(v+8*half)*64 + 32 + 16*j + lm] = (_Float16)fmaxf(c[v], 0.f);
    }
    asm volatile("" ::: "memory");

    // A operands: xc k-tile 1 and h
    v16h a1 = ldv16(&xcw[lm*64 + 32 + 16*half], &xcw[lm*64 + 32 + 16*half + 8]);
    v16h ah = ldv16(&hw [lm*32 +      16*half], &hw [lm*32 + 16*half + 8]);

    // ---- gates = xc@Wx2h^T + h@Wh2h^T + (b_x2h+b_h2h), 8 N-tiles of 16 ----
    v8f acc[8];
    #pragma unroll
    for (int j=0; j<8; ++j){
      const int n = 16*j + lm;
      float cb = sBG[n];
      v8f c = {cb,cb,cb,cb,cb,cb,cb,cb};
      v16h b0 = ldv16(&sWx2h[n*64 +      16*half], &sWx2h[n*64 +      16*half + 8]);
      v16h b1 = ldv16(&sWx2h[n*64 + 32 + 16*half], &sWx2h[n*64 + 32 + 16*half + 8]);
      v16h bh = ldv16(&sWh2h[n*32 +      16*half], &sWh2h[n*32 +      16*half + 8]);
      c = __builtin_amdgcn_wmma_f32_16x16x32_f16(false, a0, false, b0, (short)0, c, false, false);
      c = __builtin_amdgcn_wmma_f32_16x16x32_f16(false, a1, false, b1, (short)0, c, false, false);
      c = __builtin_amdgcn_wmma_f32_16x16x32_f16(false, ah, false, bh, (short)0, c, false, false);
      acc[j] = c;
    }

    // ---- LSTM pointwise; hy -> LDS (f16); output-projection partials ----
    v8f pv;
    #pragma unroll
    for (int v=0; v<8; ++v) pv[v]=0.f;

    #pragma unroll
    for (int nt=0; nt<2; ++nt){
      v8f ig = acc[nt], fg = acc[2+nt], gg = acc[4+nt], og = acc[6+nt];
      float wo = sWout[16*nt + lm];
      #pragma unroll
      for (int v=0; v<8; ++v){
        float cs = (nt==0) ? cst0[v] : cst1[v];
        cs = cs*sigm(fg[v]) + sigm(ig[v])*tanhfast(gg[v]);
        float hy = sigm(og[v])*tanhfast(cs);
        if (nt==0) cst0[v]=cs; else cst1[v]=cs;
        hw[(v+8*half)*32 + 16*nt + lm] = (_Float16)hy;
        pv[v] += hy*wo;
      }
    }

    // ---- allreduce over each 16-lane half: xn_next[M = v + 8*half] ----
    #pragma unroll
    for (int v=0; v<8; ++v){
      float s = pv[v];
      s += __shfl_xor(s, 1, 32);
      s += __shfl_xor(s, 2, 32);
      s += __shfl_xor(s, 4, 32);
      s += __shfl_xor(s, 8, 32);
      pv[v] = s;
    }
    if (lm == 0){
      #pragma unroll
      for (int v=0; v<8; ++v) sXN[w][8*half + v] = pv[v] + bout;
    }
    asm volatile("" ::: "memory");
    xnv = sXN[w][lm];                        // feed back as next-step input
    if (half == 0)
      out[bidx*(Tsz*Nsz) + t*Nsz + nidx] = xnv;  // preds[b, t, n]
  }
}

extern "C" void kernel_launch(void* const* d_in, const int* in_sizes, int n_in,
                              void* d_out, int out_size, void* d_ws, size_t ws_size,
                              hipStream_t stream) {
  (void)in_sizes; (void)n_in; (void)d_ws; (void)ws_size; (void)out_size;
  const float* x     = (const float*)d_in[0];
  // d_in[1..3] = A_q, A_h, A : dead inputs (ChebConv K=1 ignores the graph)
  const float* W_in  = (const float*)d_in[4];
  const float* b_in  = (const float*)d_in[5];
  const float* W_gc  = (const float*)d_in[6];
  const float* b_gc  = (const float*)d_in[7];
  const float* W_x2h = (const float*)d_in[8];
  const float* b_x2h = (const float*)d_in[9];
  const float* W_h2h = (const float*)d_in[10];
  const float* b_h2h = (const float*)d_in[11];
  const float* W_out = (const float*)d_in[12];
  const float* b_out = (const float*)d_in[13];
  float* out = (float*)d_out;

  dim3 grid(ROWS / ROWS_PER_BLOCK);   // 512 blocks x 256 threads = 4096 waves x 16 rows
  dim3 block(256);
  hipLaunchKernelGGL(gc_lstm_kernel, grid, block, 0, stream,
                     x, W_in, b_in, W_gc, b_gc, W_x2h, b_x2h, W_h2h, b_h2h,
                     W_out, b_out, out);
}
